// QConv2d_47923245089350
// MI455X (gfx1250) — compile-verified, matrix operand order checked by reading, not run on hardware
//
#include <hip/hip_runtime.h>

typedef __attribute__((ext_vector_type(16))) _Float16 v16h;
typedef __attribute__((ext_vector_type(8)))  float    v8f;
typedef __attribute__((ext_vector_type(4)))  unsigned int u32x4;
typedef __attribute__((ext_vector_type(8)))  int      i32x8;
typedef __attribute__((ext_vector_type(4)))  int      i32x4;

union Frag16 {
    u32x4    q[2];
    v16h     h;
    _Float16 e[16];
};

#define QSCALE     128.0f
#define INV_QSCALE 0.0078125f
#define LDSTRIDE   40          // f16 per (h,w) halo cell: 80B = 64B data + 16B pad

// Per-lane K pattern within a 32-wide K block for 16-bit WMMA operands.
__device__ __forceinline__ int kpat(int e, int h) {
    return (e & 7) + 8 * h + 16 * (e >> 3);
}

// ---------------------------------------------------------------------------
// Prep 1: quantize W (round(W*128)) to f16, pre-swizzled into WMMA A-fragment
// order: wf[kb][ocb][lane][e], kb = cblk*9 + r. 32B contiguous per lane.
// ---------------------------------------------------------------------------
__global__ __launch_bounds__(256) void pack_weights(const float* __restrict__ W,
                                                    _Float16* __restrict__ wf) {
    int id = blockIdx.x * blockDim.x + threadIdx.x;
    if (id >= 36 * 16 * 32) return;
    int lane = id & 31;
    int ocb  = (id >> 5) & 15;
    int kb   = id >> 9;
    int cblk = kb / 9;
    int r    = kb % 9;
    int kh   = r / 3, kw = r % 3;
    int oc   = ocb * 16 + (lane & 15);
    int half = lane >> 4;
    Frag16 f;
    #pragma unroll
    for (int e = 0; e < 16; ++e) {
        int c   = cblk * 32 + kpat(e, half);
        float w = W[((oc * 128 + c) * 3 + kh) * 3 + kw];
        f.e[e]  = (_Float16)rintf(w * QSCALE);
    }
    u32x4* dst = (u32x4*)wf + (size_t)id * 2;
    dst[0] = f.q[0];
    dst[1] = f.q[1];
}

// ---------------------------------------------------------------------------
// Prep 2: x (NCHW f32) -> xh (N, 58, 66, 128) NHWC f16 with zero border
// (padding pre-materialized so the hot loop / TDM needs no bounds logic).
// LDS transpose tile keeps both global reads and writes coalesced.
// ---------------------------------------------------------------------------
__global__ __launch_bounds__(256) void x_to_nhwc_f16(const float* __restrict__ x,
                                                     _Float16* __restrict__ xh) {
    __shared__ _Float16 t[32 * 136];      // [w_local][c], stride padded vs banks
    const int h   = blockIdx.x;           // 0..55
    const int n   = blockIdx.y;           // 0..31
    const int tid = threadIdx.x;
    for (int wblk = 0; wblk < 2; ++wblk) {
        __syncthreads();
        const int wl = tid & 31;
        const int w  = wblk * 32 + wl;
        #pragma unroll
        for (int i = 0; i < 16; ++i) {    // reads coalesced along w
            int c   = (tid >> 5) * 16 + i;
            float v = (w < 56) ? x[(((long)n * 128 + c) * 56 + h) * 56 + w] : 0.0f;
            t[wl * 136 + c] = (_Float16)v;
        }
        __syncthreads();
        #pragma unroll
        for (int i = 0; i < 16; ++i) {    // writes coalesced along c
            int idx = tid + 256 * i;
            int w2  = idx >> 7;
            int c   = idx & 127;
            if (wblk * 32 + w2 < 56)
                xh[(((long)n * 58 + h + 1) * 66 + (wblk * 32 + w2 + 1)) * 128 + c] =
                    t[w2 * 136 + c];
        }
    }
}

// ---------------------------------------------------------------------------
// TDM descriptor issue: 3D tile (32 c contiguous, 18 w, 10 h) from padded NHWC
// f16 tensor into LDS, with pad_amount=4dw after every pad_interval=16dw so
// each 64B channel-row lands on an 80B LDS cell stride (bank-friendly, and
// 16B-aligned for the ds_load_b128 fragment reads).
// ---------------------------------------------------------------------------
__device__ __forceinline__ void tdm_load_halo(const _Float16* xh, unsigned lds_off,
                                              int n, int hbase, int wbase, int cblk) {
    unsigned long long ga = (unsigned long long)(uintptr_t)xh +
        2ull * (((((unsigned long long)n * 58 + hbase) * 66) + wbase) * 128 + cblk * 32);
    u32x4 g0;
    g0[0] = 1u;                                   // count=1 valid descriptor
    g0[1] = lds_off;                              // LDS byte address
    g0[2] = (unsigned)ga;                         // global_addr[31:0]
    g0[3] = ((unsigned)(ga >> 32) & 0x01FFFFFFu)  // global_addr[56:32]
            | 0x80000000u;                        // type=2 ("image")
    i32x8 g1;
    g1[0] = (1 << 16)                             // data_size = 2 bytes
          | (1 << 20)                             // pad_enable
          | (3 << 22)                             // pad_interval: 16 dwords
          | (3 << 25);                            // pad_amount:   4 dwords
    g1[1] = (128 << 16);                          // tensor_dim0 (c) low16
    g1[2] = (66 << 16);                           // dim0 hi | tensor_dim1 (w) low16
    g1[3] = (32 << 16);                           // dim1 hi | tile_dim0 = 32 c
    g1[4] = 18 | (10 << 16);                      // tile_dim1 = 18 w, tile_dim2 = 10 h
    g1[5] = 128;                                  // tensor_dim0_stride (w step, elems)
    g1[6] = (8448 << 16);                         // stride0 hi | tensor_dim1_stride lo (66*128)
    g1[7] = 0;                                    // tensor_dim1_stride hi
    i32x4 g2;
    g2[0] = 58;                                   // tensor_dim2 (h)
    g2[1] = 0; g2[2] = 0; g2[3] = 0;              // no dim3, no iterate
    i32x4 g3 = {0, 0, 0, 0};
    i32x8 g4 = {0, 0, 0, 0, 0, 0, 0, 0};          // unused extension group (clang-23 form)
    __builtin_amdgcn_tensor_load_to_lds(g0, g1, g2, g3, g4, 0);
}

// ---------------------------------------------------------------------------
// Main (TDM path): implicit-GEMM conv, double-buffered LDS halo tiles fed by
// the Tensor Data Mover, fully overlapped with WMMA compute.
// Block tile = 128 oc x (8h x 16w); 8 waves: wm->32 oc, wn->4 rows of 16 w.
// ---------------------------------------------------------------------------
__global__ __launch_bounds__(256) void qconv_wmma_tdm(const _Float16* __restrict__ xh,
                                                      const _Float16* __restrict__ wf,
                                                      const float* __restrict__ bias,
                                                      float* __restrict__ out) {
    __shared__ _Float16 lds[2][10 * 18 * LDSTRIDE];

    const int tid  = threadIdx.x;
    const int lane = tid & 31;
    const int wv   = tid >> 5;
    const int wm   = wv & 3;
    const int wn   = wv >> 2;
    const int l16  = lane & 15;
    const int half = lane >> 4;

    const int bw    = blockIdx.x & 3;
    const int bh    = blockIdx.x >> 2;
    const int ocT   = blockIdx.y * 128;
    const int n     = blockIdx.z;
    const int hbase = bh * 8;
    const int wbase = bw * 16;

    const unsigned ldsb[2] = { (unsigned)(size_t)&lds[0][0],
                               (unsigned)(size_t)&lds[1][0] };

    v8f acc[2][4] = {};

    if (wv == 0) tdm_load_halo(xh, ldsb[0], n, hbase, wbase, 0);

    for (int cblk = 0; cblk < 4; ++cblk) {
        if (wv == 0) __builtin_amdgcn_s_wait_tensorcnt(0);  // tile cblk landed
        __syncthreads();                                    // publish to all waves
        if (wv == 0 && cblk < 3)                            // prefetch next tile into
            tdm_load_halo(xh, ldsb[(cblk + 1) & 1],         // the buffer everyone just
                          n, hbase, wbase, cblk + 1);       // finished reading
        const _Float16* tile = &lds[cblk & 1][0];

        for (int r = 0; r < 9; ++r) {
            const int kh = r / 3, kw = r % 3;
            const int kb = cblk * 9 + r;

            if (kb + 1 < 36)
                __builtin_prefetch((const char*)wf +
                    (size_t)((kb + 1) * 16 + blockIdx.y * 8 + wm * 2) * 32 * 32, 0, 0);

            Frag16 afrag[2];
            #pragma unroll
            for (int fa = 0; fa < 2; ++fa) {
                int ocb = blockIdx.y * 8 + wm * 2 + fa;
                const u32x4* src =
                    (const u32x4*)wf + ((size_t)(kb * 16 + ocb) * 32 + lane) * 2;
                afrag[fa].q[0] = src[0];
                afrag[fa].q[1] = src[1];
            }

            #pragma unroll
            for (int fb = 0; fb < 4; ++fb) {
                int mh   = wn * 4 + fb;
                int cell = (mh + kh) * 18 + (l16 + kw);
                const u32x4* p = (const u32x4*)&tile[cell * LDSTRIDE + half * 8];
                Frag16 bfrag;
                bfrag.q[0] = p[0];
                bfrag.q[1] = p[2];
                #pragma unroll
                for (int fa = 0; fa < 2; ++fa) {
                    acc[fa][fb] = __builtin_amdgcn_wmma_f32_16x16x32_f16(
                        false, afrag[fa].h, false, bfrag.h,
                        (short)0, acc[fa][fb], false, false);
                }
            }
        }
        __syncthreads();   // all reads of this buffer done before next overwrite
    }

    const int ow = wbase + l16;
    if (ow < 56) {
        #pragma unroll
        for (int fb = 0; fb < 4; ++fb) {
            int oh = hbase + wn * 4 + fb;
            #pragma unroll
            for (int fa = 0; fa < 2; ++fa) {
                int ocbase = ocT + wm * 32 + fa * 16 + half * 8;
                #pragma unroll
                for (int v = 0; v < 8; ++v) {
                    int oc   = ocbase + v;
                    float bq = rintf(bias[oc] * QSCALE);
                    out[(((long)n * 256 + oc) * 56 + oh) * 56 + ow] =
                        acc[fa][fb][v] * INV_QSCALE + bq;
                }
            }
        }
    }
}

// ---------------------------------------------------------------------------
// Fallback: direct cooperative halo fill from f32 NCHW x.
// Used only if ws can't hold the padded NHWC f16 tensor.
// ---------------------------------------------------------------------------
__global__ __launch_bounds__(256) void qconv_wmma_direct(const float* __restrict__ x,
                                                         const _Float16* __restrict__ wf,
                                                         const float* __restrict__ bias,
                                                         float* __restrict__ out) {
    __shared__ _Float16 lds[10 * 18 * LDSTRIDE];

    const int tid  = threadIdx.x;
    const int lane = tid & 31;
    const int wv   = tid >> 5;
    const int wm   = wv & 3;
    const int wn   = wv >> 2;
    const int l16  = lane & 15;
    const int half = lane >> 4;

    const int bw    = blockIdx.x & 3;
    const int bh    = blockIdx.x >> 2;
    const int ocT   = blockIdx.y * 128;
    const int n     = blockIdx.z;
    const int hbase = bh * 8;
    const int wbase = bw * 16;

    v8f acc[2][4] = {};

    for (int cblk = 0; cblk < 4; ++cblk) {
        __syncthreads();
        for (int idx = tid; idx < 10 * 18 * 32; idx += 256) {
            int c   = idx / 180;
            int pos = idx % 180;
            int hh  = pos / 18;
            int ww  = pos % 18;
            int gh  = hbase + hh - 1;
            int gw  = wbase + ww - 1;
            float v = 0.0f;
            if (gh >= 0 && gh < 56 && gw >= 0 && gw < 56)
                v = x[(((long)n * 128 + cblk * 32 + c) * 56 + gh) * 56 + gw];
            lds[(hh * 18 + ww) * LDSTRIDE + c] = (_Float16)v;
        }
        __syncthreads();

        for (int r = 0; r < 9; ++r) {
            const int kh = r / 3, kw = r % 3;
            const int kb = cblk * 9 + r;
            Frag16 afrag[2];
            #pragma unroll
            for (int fa = 0; fa < 2; ++fa) {
                int ocb = blockIdx.y * 8 + wm * 2 + fa;
                const u32x4* src =
                    (const u32x4*)wf + ((size_t)(kb * 16 + ocb) * 32 + lane) * 2;
                afrag[fa].q[0] = src[0];
                afrag[fa].q[1] = src[1];
            }
            #pragma unroll
            for (int fb = 0; fb < 4; ++fb) {
                int mh   = wn * 4 + fb;
                int cell = (mh + kh) * 18 + (l16 + kw);
                const u32x4* p = (const u32x4*)&lds[cell * LDSTRIDE + half * 8];
                Frag16 bfrag;
                bfrag.q[0] = p[0];
                bfrag.q[1] = p[2];
                #pragma unroll
                for (int fa = 0; fa < 2; ++fa) {
                    acc[fa][fb] = __builtin_amdgcn_wmma_f32_16x16x32_f16(
                        false, afrag[fa].h, false, bfrag.h,
                        (short)0, acc[fa][fb], false, false);
                }
            }
        }
    }

    const int ow = wbase + l16;
    if (ow < 56) {
        #pragma unroll
        for (int fb = 0; fb < 4; ++fb) {
            int oh = hbase + wn * 4 + fb;
            #pragma unroll
            for (int fa = 0; fa < 2; ++fa) {
                int ocbase = ocT + wm * 32 + fa * 16 + half * 8;
                #pragma unroll
                for (int v = 0; v < 8; ++v) {
                    int oc   = ocbase + v;
                    float bq = rintf(bias[oc] * QSCALE);
                    out[(((long)n * 256 + oc) * 56 + oh) * 56 + ow] =
                        acc[fa][fb][v] * INV_QSCALE + bq;
                }
            }
        }
    }
}

extern "C" void kernel_launch(void* const* d_in, const int* in_sizes, int n_in,
                              void* d_out, int out_size, void* d_ws, size_t ws_size,
                              hipStream_t stream) {
    const float* x = (const float*)d_in[0];   // (32,128,56,56)
    const float* W = (const float*)d_in[1];   // (256,128,3,3)
    const float* b = (const float*)d_in[2];   // (256,)
    float* out     = (float*)d_out;           // (32,256,56,56)

    const size_t XH_OFF   = 1u << 20;                              // 1 MB aligned
    const size_t XH_BYTES = (size_t)32 * 58 * 66 * 128 * 2;        // ~31 MB

    _Float16* wf = (_Float16*)d_ws;
    pack_weights<<<(36 * 16 * 32 + 255) / 256, 256, 0, stream>>>(W, wf);

    dim3 grid(4 * 7, 2, 32);   // (w-tiles*h-tiles, oc-tiles, batch)

    if (ws_size >= XH_OFF + XH_BYTES) {
        _Float16* xh = (_Float16*)((char*)d_ws + XH_OFF);
        (void)hipMemsetAsync(xh, 0, XH_BYTES, stream);             // zero border
        dim3 tg(56, 32);
        x_to_nhwc_f16<<<tg, 256, 0, stream>>>(x, xh);
        qconv_wmma_tdm<<<grid, 256, 0, stream>>>(xh, wf, b, out);
    } else {
        qconv_wmma_direct<<<grid, 256, 0, stream>>>(x, wf, b, out);
    }
}
